// Block_16071767622365
// MI455X (gfx1250) — matrix operand order repaired and checked by reading, not compiled, not run on hardware
//
#include <hip/hip_runtime.h>
#include <hip/hip_bf16.h>

// ---------------------------------------------------------------------------
// Problem constants (match reference)
// ---------------------------------------------------------------------------
constexpr int Bc  = 2;
constexpr int Tc  = 2048;
constexpr int Dc  = 2048;
constexpr int Hc  = 16;
constexpr int DHc = 128;
constexpr int FGc = 5461;
constexpr int FGPc = 5504;          // FG padded to multiple of 128 (tile-aligned)
constexpr int BTc = Bc * Tc;        // 4096 rows
constexpr float EPSc = 1e-5f;

// ---------------------------------------------------------------------------
// CDNA5 async global->LDS support (guarded; falls back to cooperative staging)
// ---------------------------------------------------------------------------
#if defined(__has_builtin)
#if __has_builtin(__builtin_amdgcn_global_load_async_to_lds_b128)
#define HAVE_ASYNC_LDS 1
#endif
#endif
#ifndef HAVE_ASYNC_LDS
#define HAVE_ASYNC_LDS 0
#endif

typedef int v4i __attribute__((ext_vector_type(4)));
typedef __attribute__((address_space(1))) v4i gv4i;   // global int4
typedef __attribute__((address_space(3))) v4i lv4i;   // LDS int4

// ---------------------------------------------------------------------------
// WMMA types + helpers (CDNA5 wave32, v_wmma_f32_16x16x32_bf16)
// ---------------------------------------------------------------------------
typedef __attribute__((ext_vector_type(16))) __bf16 v16bf;
typedef __attribute__((ext_vector_type(8)))  __bf16 v8bf;
typedef __attribute__((ext_vector_type(8)))  float  v8f;

__device__ __forceinline__ v16bf combine16(v8bf lo, v8bf hi) {
  v16bf r;
#pragma unroll
  for (int i = 0; i < 8; ++i) { r[i] = lo[i]; r[8 + i] = hi[i]; }
  return r;
}

// A-operand: 16(M) x 32(K) tile of row-major [*, lda] bf16.
// ISA layout: lanes 0-15 hold M=lane, K = {k0..k0+7, k0+16..k0+23};
//             lanes 16-31 hold M=lane-16, K = {k0+8..k0+15, k0+24..k0+31}.
__device__ __forceinline__ v16bf load_a_frag(const __bf16* __restrict__ A, int lda,
                                             int row0, int k0, int lane) {
  const int r  = row0 + (lane & 15);
  const int hi = (lane >> 4) & 1;
  const __bf16* p = A + (size_t)r * lda + k0 + hi * 8;
  v8bf lo = *(const v8bf*)p;          // K = k0 + hi*8 + [0..7]
  v8bf hh = *(const v8bf*)(p + 16);   // K = k0 + 16 + hi*8 + [0..7]
  return combine16(lo, hh);
}

// B-operand: 32(K) x 16(N) tile, source W[N,K] row-major (K contiguous).
// ISA layout: lanes 0-15 hold column N=lane with K = {k0..k0+7, k0+16..k0+23};
//             lanes 16-31 hold N=lane-16 with K = {k0+8..k0+15, k0+24..k0+31}.
__device__ __forceinline__ v16bf load_b_frag(const __bf16* __restrict__ W, int ldb,
                                             int n0, int k0, int lane) {
  const int n  = n0 + (lane & 15);
  const int hi = (lane >> 4) & 1;
  const __bf16* p = W + (size_t)n * ldb + k0 + hi * 8;
  v8bf lo = *(const v8bf*)p;
  v8bf hh = *(const v8bf*)(p + 16);
  return combine16(lo, hh);
}

__device__ __forceinline__ v8f wmma_bf16(v16bf a, v16bf b, v8f c) {
  return __builtin_amdgcn_wmma_f32_16x16x32_bf16(
      /*neg_a=*/false, a, /*neg_b=*/false, b,
      /*c_mod=*/(short)0, c, /*reuse_a=*/false, /*reuse_b=*/false);
}

__device__ __forceinline__ float hsum16(float v) {
#pragma unroll
  for (int m = 8; m >= 1; m >>= 1) v += __shfl_xor(v, m, 32);
  return v;
}
__device__ __forceinline__ float hmax16(float v) {
#pragma unroll
  for (int m = 8; m >= 1; m >>= 1) v = fmaxf(v, __shfl_xor(v, m, 32));
  return v;
}

// ---------------------------------------------------------------------------
// Weight convert f32 -> bf16 with zero padding to (Rp, Cp)
// ---------------------------------------------------------------------------
__global__ __launch_bounds__(256) void cvt_pad_kernel(
    const float* __restrict__ in, __bf16* __restrict__ out,
    int R, int C, int Rp, int Cp) {
  size_t i = (size_t)blockIdx.x * 256 + threadIdx.x;
  size_t total = (size_t)Rp * Cp;
  if (i >= total) return;
  int c = (int)(i % Cp);
  int r = (int)(i / Cp);
  float v = (r < R && c < C) ? in[(size_t)r * C + c] : 0.0f;
  out[i] = (__bf16)v;
}

// ---------------------------------------------------------------------------
// RMSNorm: one 256-thread block per row of [BT, D]; bf16 output
// ---------------------------------------------------------------------------
__global__ __launch_bounds__(256) void rmsnorm_bf16_kernel(
    const float* __restrict__ x, const float* __restrict__ w,
    __bf16* __restrict__ out) {
  const int row = blockIdx.x;
  const float* xr = x + (size_t)row * Dc;
  float ss = 0.0f;
  for (int i = threadIdx.x; i < Dc; i += 256) { float v = xr[i]; ss += v * v; }
#pragma unroll
  for (int m = 16; m >= 1; m >>= 1) ss += __shfl_xor(ss, m, 32);
  __shared__ float red[8];
  if ((threadIdx.x & 31) == 0) red[threadIdx.x >> 5] = ss;
  __syncthreads();
  float tot = 0.0f;
#pragma unroll
  for (int i = 0; i < 8; ++i) tot += red[i];
  const float rms = rsqrtf(tot / (float)Dc + EPSc);
  for (int i = threadIdx.x; i < Dc; i += 256)
    out[(size_t)row * Dc + i] = (__bf16)(xr[i] * rms * w[i]);
}

// ---------------------------------------------------------------------------
// WMMA GEMM: C[M,N](f32) = A[M,K](bf16) * W[N,K]^T(bf16) (+ Res)
// 8 waves; wave tile 32x64; block tile 128(M) x 128(N); K step 32.
// B block tile (128x32 = 8KB) staged through LDS, double-buffered, using
// GLOBAL_LOAD_ASYNC_TO_LDS_B128 (ASYNCcnt) when available.
// Per-wave intensity: 8 WMMA (131 KFLOP) per 2 A-frags + LDS reads.
// ---------------------------------------------------------------------------
template <bool HAS_RES>
__global__ __launch_bounds__(256) void gemm_bf16_kernel(
    const __bf16* __restrict__ A, const __bf16* __restrict__ W,
    float* __restrict__ C, const float* __restrict__ Res,
    int K, int lda, int ldb, int ldc) {
  __shared__ __bf16 bsh[2][128 * 32];   // 2 x 8KB B tiles

  const int tid  = threadIdx.x;
  const int lane = tid & 31;
  const int wave = tid >> 5;
  const int cl = lane & 15, hi = (lane >> 4) & 1;
  const int row0 = blockIdx.y * 128 + (wave >> 1) * 32;  // wave M base
  const int colB = blockIdx.x * 128;                     // block N base
  const int wn   = wave & 1;                             // wave N half

  // Staging role of this thread: copy 32B (16 elems) of one B row per K-step.
  const int srow  = tid >> 1;        // 0..127 : local B row (N)
  const int shalf = tid & 1;         // which 16-element half of the 32-elem row
  const __bf16* gsrc0 = W + (size_t)(colB + srow) * ldb + shalf * 16;

  auto stage = [&](int buf, int k0) {
    const __bf16* g = gsrc0 + k0;
    __bf16* l = &bsh[buf][srow * 32 + shalf * 16];
#if HAVE_ASYNC_LDS
    __builtin_amdgcn_global_load_async_to_lds_b128(
        (gv4i*)(g),     (lv4i*)(l),     0, 0);
    __builtin_amdgcn_global_load_async_to_lds_b128(
        (gv4i*)(g + 8), (lv4i*)(l + 8), 0, 0);
#else
    *(v8bf*)(l)     = *(const v8bf*)(g);
    *(v8bf*)(l + 8) = *(const v8bf*)(g + 8);
#endif
  };

  v8f acc[2][4];
#pragma unroll
  for (int mi = 0; mi < 2; ++mi)
#pragma unroll
    for (int ni = 0; ni < 4; ++ni)
#pragma unroll
      for (int r = 0; r < 8; ++r) acc[mi][ni][r] = 0.0f;

  const int nsteps = K / 32;
  stage(0, 0);
  for (int s = 0; s < nsteps; ++s) {
    const int k0 = s * 32;
    const int cur = s & 1;
    if (s + 1 < nsteps) {
      stage(cur ^ 1, k0 + 32);
#if HAVE_ASYNC_LDS
      asm volatile("s_wait_asynccnt 0x2" ::: "memory");  // current buf complete
#endif
    } else {
#if HAVE_ASYNC_LDS
      asm volatile("s_wait_asynccnt 0x0" ::: "memory");
#endif
    }
    __syncthreads();   // whole B tile visible to all waves

    v16bf a0 = load_a_frag(A, lda, row0,      k0, lane);
    v16bf a1 = load_a_frag(A, lda, row0 + 16, k0, lane);
    if (s + 2 < nsteps)  // keep the A stream warm (global_prefetch_b8)
      __builtin_prefetch(A + (size_t)(row0 + cl) * lda + k0 + 64, 0, 1);

    const __bf16* bs = bsh[cur];
#pragma unroll
    for (int ni = 0; ni < 4; ++ni) {
      const int nl = wn * 64 + ni * 16 + cl;           // local N column
      const __bf16* p = bs + nl * 32 + hi * 8;         // K = hi*8+[0..7], 16+hi*8+[0..7]
      v16bf b = combine16(*(const v8bf*)p, *(const v8bf*)(p + 16));
      acc[0][ni] = wmma_bf16(a0, b, acc[0][ni]);
      acc[1][ni] = wmma_bf16(a1, b, acc[1][ni]);
    }
    __syncthreads();   // done reading buf[cur] before it is overwritten
  }

  // Epilogue: C-layout store (+ optional residual)
#pragma unroll
  for (int mi = 0; mi < 2; ++mi)
#pragma unroll
    for (int ni = 0; ni < 4; ++ni)
#pragma unroll
      for (int r = 0; r < 8; ++r) {
        const int row = row0 + mi * 16 + hi * 8 + r;
        const int col = colB + wn * 64 + ni * 16 + cl;
        const size_t idx = (size_t)row * ldc + col;
        float v = acc[mi][ni][r];
        if (HAS_RES) v += Res[idx];
        C[idx] = v;
      }
}

// ---------------------------------------------------------------------------
// RoPE + QKV split. qkv: [B,T,3,H,DH] f32 -> Q,K: [B,H,T,DH] bf16 (rotated),
//                                           Vt: [B,H,DH,T] bf16 (transposed).
// ---------------------------------------------------------------------------
__global__ __launch_bounds__(256) void rope_split_kernel(
    const float* __restrict__ qkv, __bf16* __restrict__ Qb,
    __bf16* __restrict__ Kb, __bf16* __restrict__ Vt) {
  const int PH = DHc / 2;  // 64
  size_t idx = (size_t)blockIdx.x * 256 + threadIdx.x;
  size_t total = (size_t)Bc * Hc * Tc * PH;
  if (idx >= total) return;
  const int j = (int)(idx % PH);
  const int t = (int)((idx / PH) % Tc);
  const int h = (int)((idx / ((size_t)PH * Tc)) % Hc);
  const int b = (int)(idx / ((size_t)PH * Tc * Hc));

  const size_t base = ((size_t)(b * Tc + t)) * (3 * Dc) + (size_t)h * DHc + 2 * j;
  const float q0 = qkv[base],          q1 = qkv[base + 1];
  const float k0 = qkv[base + Dc],     k1 = qkv[base + Dc + 1];
  const float v0 = qkv[base + 2 * Dc], v1 = qkv[base + 2 * Dc + 1];

  const float invf = powf(10000.0f, -((float)(2 * j)) / (float)DHc);
  const float ang = (float)t * invf;
  const float c = cosf(ang), s = sinf(ang);

  const int bh = b * Hc + h;
  const size_t qo = ((size_t)bh * Tc + t) * DHc + 2 * j;
  Qb[qo]     = (__bf16)(q0 * c - q1 * s);
  Qb[qo + 1] = (__bf16)(q1 * c + q0 * s);
  Kb[qo]     = (__bf16)(k0 * c - k1 * s);
  Kb[qo + 1] = (__bf16)(k1 * c + k0 * s);

  const size_t vo = (size_t)bh * DHc * Tc;
  Vt[vo + (size_t)(2 * j) * Tc + t]     = (__bf16)v0;
  Vt[vo + (size_t)(2 * j + 1) * Tc + t] = (__bf16)v1;
}

// ---------------------------------------------------------------------------
// Flash attention (causal). One wave per 16-query tile, 8 waves/block.
// Q,K: [B,H,T,DH] bf16 ; Vt: [B,H,DH,T] bf16 ; Y: [B*T, D] bf16.
// Key blocks of 32; online softmax; S and O via v_wmma_f32_16x16x32_bf16.
// No __syncthreads in the loop (waves have unequal causal trip counts).
// ---------------------------------------------------------------------------
__global__ __launch_bounds__(256) void flash_attn_kernel(
    const __bf16* __restrict__ Q, const __bf16* __restrict__ Kb,
    const __bf16* __restrict__ Vt, __bf16* __restrict__ Y) {
  __shared__ __bf16 pls[8][16 * 32];   // per-wave P tile staging (8 KB)

  const int lane = threadIdx.x & 31;
  const int wave = threadIdx.x >> 5;
  const int cl = lane & 15, hi = (lane >> 4) & 1;
  const int bh = blockIdx.y;
  const int b = bh >> 4, h = bh & 15;
  const int m0 = blockIdx.x * 128 + wave * 16;

  const __bf16* Qh = Q  + (size_t)bh * Tc * DHc;
  const __bf16* Kh = Kb + (size_t)bh * Tc * DHc;
  const __bf16* Vh = Vt + (size_t)bh * DHc * Tc;

  // Preload the 4 A-fragments of this wave's Q tile (DH=128 = 4 * K32).
  v16bf qa[4];
  {
    const __bf16* qr = Qh + (size_t)(m0 + cl) * DHc;
#pragma unroll
    for (int c = 0; c < 4; ++c) {
      const int d0 = 32 * c;
      v8bf lo = *(const v8bf*)(qr + d0 + hi * 8);
      v8bf hh = *(const v8bf*)(qr + d0 + 16 + hi * 8);
      qa[c] = combine16(lo, hh);
    }
  }

  v8f o[8];
#pragma unroll
  for (int nt = 0; nt < 8; ++nt)
#pragma unroll
    for (int r = 0; r < 8; ++r) o[nt][r] = 0.0f;

  float mst[8], lst[8];
#pragma unroll
  for (int r = 0; r < 8; ++r) { mst[r] = -1e30f; lst[r] = 0.0f; }

  const float scale = 0.0883883476483f;  // 1/sqrt(128)

  for (int n0 = 0; n0 <= m0 + 15; n0 += 32) {
    // ---- S = Q K^T for a 16x32 score tile (two 16x16 accumulators) ----
    v8f s0, s1;
#pragma unroll
    for (int r = 0; r < 8; ++r) { s0[r] = 0.0f; s1[r] = 0.0f; }
#pragma unroll
    for (int c = 0; c < 4; ++c) {
      const int d0 = 32 * c;
      v16bf kb0 = load_b_frag(Kh, DHc, n0,      d0, lane);
      s0 = wmma_bf16(qa[c], kb0, s0);
      v16bf kb1 = load_b_frag(Kh, DHc, n0 + 16, d0, lane);
      s1 = wmma_bf16(qa[c], kb1, s1);
    }

    // ---- online softmax over this 32-key block ----
    const int col0 = n0 + cl, col1 = col0 + 16;
    float p0[8], p1[8], facs[8];
#pragma unroll
    for (int r = 0; r < 8; ++r) {
      const int row = m0 + hi * 8 + r;
      const bool ok0 = (col0 <= row), ok1 = (col1 <= row);
      const float v0 = ok0 ? s0[r] * scale : -1e30f;
      const float v1 = ok1 ? s1[r] * scale : -1e30f;
      const float mc = hmax16(fmaxf(v0, v1));
      const float mo = mst[r];
      const float mn = fmaxf(mo, mc);
      const float fac = (mc > mo) ? __expf(mo - mn) : 1.0f;
      const float e0 = ok0 ? __expf(v0 - mn) : 0.0f;
      const float e1 = ok1 ? __expf(v1 - mn) : 0.0f;
      lst[r] = lst[r] * fac + hsum16(e0 + e1);
      mst[r] = mn;
      facs[r] = fac;
      p0[r] = e0; p1[r] = e1;
    }
#pragma unroll
    for (int nt = 0; nt < 8; ++nt)
#pragma unroll
      for (int r = 0; r < 8; ++r) o[nt][r] *= facs[r];

    // ---- P: C-layout registers -> LDS -> A-layout fragment ----
    __bf16* pw = pls[wave];
#pragma unroll
    for (int r = 0; r < 8; ++r) {
      const int prow = hi * 8 + r;
      pw[prow * 32 + cl]      = (__bf16)p0[r];
      pw[prow * 32 + 16 + cl] = (__bf16)p1[r];
    }
    asm volatile("s_wait_dscnt 0" ::: "memory");  // same-wave LDS RAW
    v16bf pa;
    {
      const __bf16* pr = pw + cl * 32;
      v8bf lo = *(const v8bf*)(pr + hi * 8);
      v8bf hh = *(const v8bf*)(pr + 16 + hi * 8);
      pa = combine16(lo, hh);
    }

    // ---- O += P * V  (V pre-transposed: Vt[d, t], K=32 keys per wmma) ----
#pragma unroll
    for (int nt = 0; nt < 8; ++nt) {
      const int d0 = nt * 16;
      const __bf16* vp = Vh + (size_t)(d0 + cl) * Tc + n0 + hi * 8;
      v8bf lo = *(const v8bf*)vp;
      v8bf hh = *(const v8bf*)(vp + 16);
      o[nt] = wmma_bf16(pa, combine16(lo, hh), o[nt]);
    }
  }

  // ---- finalize & store: Y[b, t, h*DH + d] ----
#pragma unroll
  for (int r = 0; r < 8; ++r) lst[r] = 1.0f / lst[r];
#pragma unroll
  for (int nt = 0; nt < 8; ++nt)
#pragma unroll
    for (int r = 0; r < 8; ++r) {
      const int t = m0 + hi * 8 + r;
      const int d = nt * 16 + cl;
      Y[((size_t)b * Tc + t) * Dc + (size_t)h * DHc + d] = (__bf16)(o[nt][r] * lst[r]);
    }
}

// ---------------------------------------------------------------------------
// gv = silu(gate) * val  (f32 -> bf16); padded columns stay exactly 0.
// ---------------------------------------------------------------------------
__global__ __launch_bounds__(256) void silu_mul_kernel(
    const float* __restrict__ g, const float* __restrict__ v,
    __bf16* __restrict__ out, size_t total) {
  size_t i = (size_t)blockIdx.x * 256 + threadIdx.x;
  if (i >= total) return;
  const float gv = g[i];
  const float s = gv / (1.0f + __expf(-gv));
  out[i] = (__bf16)(s * v[i]);
}

// ---------------------------------------------------------------------------
// Launcher
// ---------------------------------------------------------------------------
extern "C" void kernel_launch(void* const* d_in, const int* in_sizes, int n_in,
                              void* d_out, int out_size, void* d_ws, size_t ws_size,
                              hipStream_t stream) {
  (void)in_sizes; (void)n_in; (void)out_size; (void)ws_size;

  const float* x      = (const float*)d_in[0];
  const float* n1w    = (const float*)d_in[1];
  const float* n2w    = (const float*)d_in[2];
  const float* w_attn = (const float*)d_in[3];  // [3D, D]
  const float* w_proj = (const float*)d_in[4];  // [D, D]
  const float* w_gate = (const float*)d_in[5];  // [FG, D]
  const float* w_val  = (const float*)d_in[6];  // [FG, D]
  const float* w_mlp  = (const float*)d_in[7];  // [D, FG]
  float* out = (float*)d_out;

  // ---- workspace arena (regions reused across phases) ----
  char* ws = (char*)d_ws;
  size_t off = 0;
  auto alloc = [&](size_t bytes) -> char* {
    char* p = ws + off;
    off = (off + bytes + 255) & ~(size_t)255;
    return p;
  };
  __bf16* wqkv_b = (__bf16*)alloc((size_t)3 * Dc * Dc * 2);
  __bf16* wproj_b= (__bf16*)alloc((size_t)Dc * Dc * 2);
  __bf16* wgate_b= (__bf16*)alloc((size_t)FGPc * Dc * 2);
  __bf16* wval_b = (__bf16*)alloc((size_t)FGPc * Dc * 2);
  __bf16* wmlp_b = (__bf16*)alloc((size_t)Dc * FGPc * 2);
  __bf16* actb   = (__bf16*)alloc((size_t)BTc * Dc * 2);        // h1 / y / h2 (reused)
  float*  bigf   = (float*) alloc((size_t)BTc * 3 * Dc * 4);    // qkv f32, later gate f32
  __bf16* qkvb   = (__bf16*)alloc((size_t)3 * Bc * Hc * Tc * DHc * 2); // q,k,vT; later gv
  float*  x1f    = (float*) alloc((size_t)BTc * Dc * 4);
  float*  valf   = (float*) alloc((size_t)BTc * FGPc * 4);

  __bf16* qb  = qkvb;
  __bf16* kb  = qkvb + (size_t)Bc * Hc * Tc * DHc;
  __bf16* vtb = qkvb + (size_t)2 * Bc * Hc * Tc * DHc;
  float*  gatef = bigf;
  __bf16* gvb   = qkvb;

  auto cvt = [&](const float* in, __bf16* o, int R, int C, int Rp, int Cp) {
    size_t total = (size_t)Rp * Cp;
    cvt_pad_kernel<<<(unsigned)((total + 255) / 256), 256, 0, stream>>>(in, o, R, C, Rp, Cp);
  };

  // 1) weights -> bf16 (FG dim zero-padded to FGPc)
  cvt(w_attn, wqkv_b, 3 * Dc, Dc, 3 * Dc, Dc);
  cvt(w_proj, wproj_b, Dc, Dc, Dc, Dc);
  cvt(w_gate, wgate_b, FGc, Dc, FGPc, Dc);
  cvt(w_val,  wval_b,  FGc, Dc, FGPc, Dc);
  cvt(w_mlp,  wmlp_b,  Dc, FGc, Dc, FGPc);

  // 2) h1 = rmsnorm(x, n1w) -> bf16
  rmsnorm_bf16_kernel<<<BTc, 256, 0, stream>>>(x, n1w, actb);

  // 3) qkv = h1 @ Wqkv^T : [4096, 6144] f32
  gemm_bf16_kernel<false><<<dim3(3 * Dc / 128, BTc / 128), 256, 0, stream>>>(
      actb, wqkv_b, bigf, nullptr, Dc, Dc, Dc, 3 * Dc);

  // 4) RoPE + split into Q,K [B,H,T,DH] and V^T [B,H,DH,T]
  {
    size_t total = (size_t)Bc * Hc * Tc * (DHc / 2);
    rope_split_kernel<<<(unsigned)((total + 255) / 256), 256, 0, stream>>>(bigf, qb, kb, vtb);
  }

  // 5) causal flash attention -> y [B*T, D] bf16 (reuses actb)
  flash_attn_kernel<<<dim3(Tc / 128, Bc * Hc), 256, 0, stream>>>(qb, kb, vtb, actb);

  // 6) x1 = x + y @ Wproj^T
  gemm_bf16_kernel<true><<<dim3(Dc / 128, BTc / 128), 256, 0, stream>>>(
      actb, wproj_b, x1f, x, Dc, Dc, Dc, Dc);

  // 7) h2 = rmsnorm(x1, n2w) -> bf16 (reuses actb)
  rmsnorm_bf16_kernel<<<BTc, 256, 0, stream>>>(x1f, n2w, actb);

  // 8) gate = h2 @ Wgate^T : [4096, FGPc] f32 (reuses bigf)
  gemm_bf16_kernel<false><<<dim3(FGPc / 128, BTc / 128), 256, 0, stream>>>(
      actb, wgate_b, gatef, nullptr, Dc, Dc, Dc, FGPc);

  // 9) val = h2 @ Wval^T
  gemm_bf16_kernel<false><<<dim3(FGPc / 128, BTc / 128), 256, 0, stream>>>(
      actb, wval_b, valf, nullptr, Dc, Dc, Dc, FGPc);

  // 10) gv = silu(gate) * val -> bf16 (reuses qkvb region)
  {
    size_t total = (size_t)BTc * FGPc;
    silu_mul_kernel<<<(unsigned)((total + 255) / 256), 256, 0, stream>>>(gatef, valf, gvb, total);
  }

  // 11) out = x1 + gv @ Wmlp^T  (K = FGPc; pad region is exact zeros)
  gemm_bf16_kernel<true><<<dim3(Dc / 128, BTc / 128), 256, 0, stream>>>(
      gvb, wmlp_b, out, x1f, FGPc, FGPc, FGPc, Dc);
}
